// Model_6390911336673
// MI455X (gfx1250) — compile-verified, matrix operand order checked by reading, not run on hardware
//
#include <hip/hip_runtime.h>
#include <hip/hip_bf16.h>
#include <stddef.h>

// ---------------------------------------------------------------------------
// MoE grouped GEMM for MI455X (gfx1250, wave32, WMMA).
//   out[m,t,:] = (A[m] @ B[e].T + bias[e]) * w[m,t],  e = topk_ids[m,t]
// M=256, K=1024, N=2048, E=64, TOPK=8 -> L=2048 routed rows, cap=32 rows/expert.
//
// Roofline: 8.6 GFLOP vs 512 MB of expert weights B read exactly once
// -> HBM-bound (~22 us @ 23.3 TB/s). Strategy: stream B straight from HBM
// (each (expert, N-tile) owned by one workgroup -> every B cacheline touched
// once), keep the expert's 32 A rows resident in LDS as bf16 (64 KB of the
// 320 KB WGP LDS), run the math on v_wmma_f32_16x16x32_bf16 with f32
// accumulation. f32->bf16 conversion is done at 1.5 VALU ops / element
// (add-round + v_perm_b32 packing two elements per op) so conversion stays
// far below the memory time.
// ---------------------------------------------------------------------------

constexpr int Mdim = 256;
constexpr int Kdim = 1024;
constexpr int Ndim = 2048;
constexpr int Edim = 64;
constexpr int TOPK = 8;
constexpr int Ldim = Mdim * TOPK;   // 2048 routed rows
constexpr int CAP  = Ldim / Edim;   // 32 rows per expert (balanced routing)
constexpr int NT   = 256;           // N-tile per workgroup -> B read exactly once

typedef __attribute__((ext_vector_type(16))) __bf16    v16bf;
typedef __attribute__((ext_vector_type(8)))  __bf16    v8bf;
typedef __attribute__((ext_vector_type(8)))  float     v8f;
typedef __attribute__((ext_vector_type(4)))  float     v4f;
typedef __attribute__((ext_vector_type(8)))  unsigned  v8u;

// Pack two f32 -> one dword of two bf16 (round-half-up + v_perm_b32 merge).
// result[15:0] = bf16(lo), result[31:16] = bf16(hi): 3 VALU ops per 2 elems.
__device__ __forceinline__ unsigned pack2_bf16(float lo, float hi) {
    unsigned a = __builtin_bit_cast(unsigned, lo) + 0x8000u;
    unsigned b = __builtin_bit_cast(unsigned, hi) + 0x8000u;
    // sel 0x07060302: bytes {b3,b2,a3,a2} -> {bf16(hi), bf16(lo)}
    return __builtin_amdgcn_perm(b, a, 0x07060302u);
}

__device__ __forceinline__ __bf16 to_bf16(float f) {
    unsigned u = __builtin_bit_cast(unsigned, f) + 0x8000u;
    unsigned short h = (unsigned short)(u >> 16);
    return __builtin_bit_cast(__bf16, h);
}

// Build a lane's 16-element bf16 WMMA fragment from 8 floats at p0 (K low
// octet) and 8 floats at p1 (K high octet), per the ISA 16-bit operand
// striping (elements 0..7 = K base+0..7, elements 8..15 = K base+16..23;
// the lane-half supplies the +8 offset).
__device__ __forceinline__ v16bf cvt_frag(const float* __restrict__ p0,
                                          const float* __restrict__ p1) {
    v4f a = *(const v4f*)(p0);
    v4f b = *(const v4f*)(p0 + 4);
    v4f c = *(const v4f*)(p1);
    v4f d = *(const v4f*)(p1 + 4);
    v8u p;
    p[0] = pack2_bf16(a[0], a[1]);
    p[1] = pack2_bf16(a[2], a[3]);
    p[2] = pack2_bf16(b[0], b[1]);
    p[3] = pack2_bf16(b[2], b[3]);
    p[4] = pack2_bf16(c[0], c[1]);
    p[5] = pack2_bf16(c[2], c[3]);
    p[6] = pack2_bf16(d[0], d[1]);
    p[7] = pack2_bf16(d[2], d[3]);
    return __builtin_bit_cast(v16bf, p);
}

// --------------------------- routing kernel --------------------------------
// Stable argsort by expert id under balanced routing: thread e scans the flat
// topk_ids in order and records its CAP slot indices.
__global__ void __launch_bounds__(64)
moe_route_kernel(const int* __restrict__ topk_ids, int* __restrict__ slots) {
    int e = threadIdx.x;
    int cnt = 0;
    for (int l = 0; l < Ldim && cnt < CAP; ++l) {
        if (topk_ids[l] == e) {
            slots[e * CAP + cnt] = l;
            ++cnt;
        }
    }
}

// ----------------------------- main kernel ---------------------------------
// grid = (E, N/NT) = (64, 8); block = 256 threads = 8 waves.
// Wave w owns C rows [16*(w&1) .. +15] and cols [64*(w>>1) .. +63] of the
// 32 x 256 block tile: 4 accumulators of 16x16 f32.
__global__ void __launch_bounds__(256)
moe_wmma_kernel(const float* __restrict__ A,
                const float* __restrict__ B,
                const float* __restrict__ bias,
                const float* __restrict__ tw,
                const int*   __restrict__ slots,
                float*       __restrict__ out) {
    __shared__ __attribute__((aligned(16))) __bf16 ldsA[CAP * Kdim]; // 64 KB

    const int e    = blockIdx.x;
    const int nb   = blockIdx.y;
    const int tid  = threadIdx.x;
    const int lane = tid & 31;
    const int wave = tid >> 5;

    // ---- stage the expert's 32 A rows into LDS as bf16 (coalesced on K) ----
    for (int i = tid; i < CAP * Kdim; i += 256) {
        int c = i >> 10;             // row within expert group
        int k = i & (Kdim - 1);
        int slot = slots[e * CAP + c];
        int m = slot >> 3;           // slot / TOPK
        ldsA[i] = to_bf16(A[(size_t)m * Kdim + k]);
    }
    __syncthreads();

    const int mtile = wave & 1;          // 0..1   (M sub-tile of 16)
    const int nsub  = (wave >> 1) << 6;  // 0,64,128,192
    const int nbase = nb * NT + nsub;
    const int half  = lane >> 4;         // lane half selects K octet group
    const int l16   = lane & 15;
    const int koff  = half * 8;

    const __bf16* aRow  = &ldsA[(size_t)(mtile * 16 + l16) * Kdim];
    const float*  bBase = B + (size_t)e * Ndim * Kdim;

    v8f acc[4];
#pragma unroll
    for (int j = 0; j < 4; ++j) acc[j] = (v8f)(0.0f);

#pragma unroll 2
    for (int kk = 0; kk < Kdim; kk += 32) {
        // A fragment: two 16B ds_load_b128 per lane, ISA operand striping.
        v8bf alo = *(const v8bf*)(aRow + kk + koff);
        v8bf ahi = *(const v8bf*)(aRow + kk + 16 + koff);
        v16bf aF = __builtin_shufflevector(alo, ahi,
                                           0, 1, 2, 3, 4, 5, 6, 7,
                                           8, 9, 10, 11, 12, 13, 14, 15);

#pragma unroll
        for (int j = 0; j < 4; ++j) {
            // B row n = nbase + j*16 + l16; the two lane-halves of a wave
            // cover 16 rows x full 128B per row per k-step -> every B
            // cacheline consumed exactly once device-wide.
            const float* bRow =
                bBase + (size_t)(nbase + (j << 4) + l16) * Kdim + kk + koff;
            v16bf bF = cvt_frag(bRow, bRow + 16);
            acc[j] = __builtin_amdgcn_wmma_f32_16x16x32_bf16(
                /*neg_a=*/false, aF, /*neg_b=*/false, bF,
                /*c_mod=*/(short)0, acc[j],
                /*reuse_a=*/false, /*reuse_b=*/false);
        }
    }

    // ---- epilogue: bias + routing weight, scatter by slot ----
    // C layout: VGPR r, lanes 0-15 -> M=r, N=l16; lanes 16-31 -> M=8+r.
    int   slotArr[8];
    float wArr[8];
#pragma unroll
    for (int r = 0; r < 8; ++r) {
        int c = mtile * 16 + half * 8 + r;
        slotArr[r] = slots[e * CAP + c];
        wArr[r]    = tw[slotArr[r]];
    }
#pragma unroll
    for (int j = 0; j < 4; ++j) {
        int n = nbase + (j << 4) + l16;
        float bi = bias[e * Ndim + n];
#pragma unroll
        for (int r = 0; r < 8; ++r) {
            out[(size_t)slotArr[r] * Ndim + n] = (acc[j][r] + bi) * wArr[r];
        }
    }
}

// ----------------------------- launcher ------------------------------------
extern "C" void kernel_launch(void* const* d_in, const int* in_sizes, int n_in,
                              void* d_out, int out_size, void* d_ws, size_t ws_size,
                              hipStream_t stream) {
    (void)in_sizes; (void)n_in; (void)out_size; (void)ws_size;
    const float* A    = (const float*)d_in[0];
    const float* B    = (const float*)d_in[1];
    const float* bias = (const float*)d_in[2];
    const float* tw   = (const float*)d_in[3];
    const int*   ids  = (const int*)  d_in[4];
    float*       out  = (float*)d_out;
    int*         slots = (int*)d_ws;   // E*CAP = 2048 ints of scratch

    hipLaunchKernelGGL(moe_route_kernel, dim3(1), dim3(64), 0, stream, ids, slots);
    hipLaunchKernelGGL(moe_wmma_kernel, dim3(Edim, Ndim / NT), dim3(256), 0, stream,
                       A, B, bias, tw, slots, out);
}
